// TransformerBlock_12283606467770
// MI455X (gfx1250) — compile-verified, matrix operand order checked by reading, not run on hardware
//
#include <hip/hip_runtime.h>

typedef __attribute__((ext_vector_type(16))) __bf16 v16bf;
typedef __attribute__((ext_vector_type(8)))  __bf16 v8bf;
typedef __attribute__((ext_vector_type(8)))  float  v8f;
typedef __attribute__((ext_vector_type(4)))  int    v4i;

constexpr int  kB  = 8;
constexpr int  kM  = 512;
constexpr int  kQ  = 512;
constexpr int  kK  = 1024;
constexpr int  kD  = 1024;
constexpr int  kH  = 16;
constexpr int  kDH = 64;
constexpr long kBQ = (long)kB * kQ;      // 4096
constexpr long kBK = (long)kB * kK;      // 8192
constexpr long kQK = (long)kQ * kK;      // 524288
constexpr int  kZH = kB * kH;            // 128
constexpr float kSCALE = 0.125f;         // 1/sqrt(64)

// CDNA5 async global->LDS copy path (ASYNCcnt-tracked), with fallback.
#if defined(__gfx1250__) && __has_builtin(__builtin_amdgcn_global_load_async_to_lds_b128) && \
    __has_builtin(__builtin_amdgcn_s_wait_asynccnt)
#define USE_ASYNC_LDS 1
#else
#define USE_ASYNC_LDS 0
#endif

#define AS1 __attribute__((address_space(1)))
#define AS3 __attribute__((address_space(3)))

__device__ __forceinline__ void async_copy16(const __bf16* g, __bf16* l) {
#if USE_ASYNC_LDS
  __builtin_amdgcn_global_load_async_to_lds_b128(
      (AS1 v4i*)(void*)const_cast<__bf16*>(g), (AS3 v4i*)(void*)l, 0, 0);
#else
  *(v8bf*)l = *(const v8bf*)g;
#endif
}

__device__ __forceinline__ void wait_async_barrier() {
#if USE_ASYNC_LDS
  __builtin_amdgcn_s_wait_asynccnt(0);
#endif
  __syncthreads();
}

// ---------------------------------------------------------------------------
// Fragment loaders: 16-bit A/B matrix layout for v_wmma_*_16x16x32_bf16.
// lane<16 holds row r=lane,  K = kb+{0..7, 16..23}
// lane>=16 holds row r=lane-16, K = kb+{8..15, 24..31}
// ---------------------------------------------------------------------------
__device__ __forceinline__ v16bf load_frag(const __bf16* p, long rowStride, int lane) {
  int half = lane >> 4;
  int r    = lane & 15;
  const __bf16* q = p + (long)r * rowStride + half * 8;
  v8bf lo = *(const v8bf*)(q);
  v8bf hi = *(const v8bf*)(q + 16);
  return __builtin_shufflevector(lo, hi, 0,1,2,3,4,5,6,7,8,9,10,11,12,13,14,15);
}

// LDS tile fragment: row stride 40 halfs (80B) -> 20-bank stride, conflict-free
__device__ __forceinline__ v16bf load_frag_lds(const __bf16* base, int lane) {
  int half = lane >> 4;
  int r    = lane & 15;
  const __bf16* q = base + r * 40 + half * 8;
  v8bf lo = *(const v8bf*)(q);
  v8bf hi = *(const v8bf*)(q + 16);
  return __builtin_shufflevector(lo, hi, 0,1,2,3,4,5,6,7,8,9,10,11,12,13,14,15);
}

__device__ __forceinline__ v8f wmma_bf16(v16bf a, v16bf b, v8f c) {
  return __builtin_amdgcn_wmma_f32_16x16x32_bf16(false, a, false, b, (short)0, c, false, false);
}

// ---------------------------------------------------------------------------
// Elementwise helpers
// ---------------------------------------------------------------------------
__global__ void ew_cvt_bf16(const float* __restrict__ src, __bf16* __restrict__ dst, long n) {
  long i = (long)blockIdx.x * blockDim.x + threadIdx.x;
  if (i < n) dst[i] = (__bf16)src[i];
}

__global__ void ew_sigmul_bf16(const float* __restrict__ t, const float* __restrict__ x,
                               __bf16* __restrict__ out, long n) {
  long i = (long)blockIdx.x * blockDim.x + threadIdx.x;
  if (i >= n) return;
  float s = 1.f / (1.f + __expf(-t[i]));
  out[i] = (__bf16)(s * x[i]);
}

__global__ void ew_gru_out(const float* __restrict__ x, const float* __restrict__ tz,
                           const float* __restrict__ th, const float* __restrict__ bg,
                           float* __restrict__ out, long n) {
  long i = (long)blockIdx.x * blockDim.x + threadIdx.x;
  if (i >= n) return;
  int c = (int)(i % kD);
  float z = 1.f / (1.f + __expf(-(tz[i] - bg[c])));
  float h = tanhf(th[i]);
  out[i] = (1.f - z) * x[i] + z * h;
}

// wvT[z, d, j] = wv[(j*B+b)*D + n*DH + d],  z = b*H+n
__global__ void transpose_wv(const __bf16* __restrict__ wv, __bf16* __restrict__ wvT) {
  long idx = (long)blockIdx.x * blockDim.x + threadIdx.x;
  long total = (long)kZH * kDH * kK;
  if (idx >= total) return;
  int j = (int)(idx % kK);
  int d = (int)((idx / kK) % kDH);
  int z = (int)(idx / ((long)kK * kDH));
  int b = z / kH, n = z % kH;
  wvT[idx] = wv[((long)j * kB + b) * kD + (long)n * kDH + d];
}

// ---------------------------------------------------------------------------
// LayerNorm over D=1024 per row, bf16 output with row remapping
// ---------------------------------------------------------------------------
__global__ void ln_rows_bf16(const float* __restrict__ X, const float* __restrict__ g,
                             const float* __restrict__ bta, __bf16* __restrict__ Y,
                             int rpb_in, int rpb_out, int rowoff) {
  int row = blockIdx.x;
  int t = threadIdx.x;
  const float* x = X + (long)row * kD;
  long orow = (long)(row / rpb_in) * rpb_out + rowoff + (row % rpb_in);
  __bf16* y = Y + orow * kD;
  __shared__ float red[256];
  float v[4];
  float s = 0.f;
#pragma unroll
  for (int u = 0; u < 4; ++u) { v[u] = x[t + 256 * u]; s += v[u]; }
  red[t] = s; __syncthreads();
  for (int o = 128; o > 0; o >>= 1) { if (t < o) red[t] += red[t + o]; __syncthreads(); }
  float mean = red[0] * (1.f / kD);
  __syncthreads();
  float s2 = 0.f;
#pragma unroll
  for (int u = 0; u < 4; ++u) { float d = v[u] - mean; s2 += d * d; }
  red[t] = s2; __syncthreads();
  for (int o = 128; o > 0; o >>= 1) { if (t < o) red[t] += red[t + o]; __syncthreads(); }
  float inv = rsqrtf(red[0] * (1.f / kD) + 1e-5f);
#pragma unroll
  for (int u = 0; u < 4; ++u) {
    int c = t + 256 * u;
    y[c] = (__bf16)((v[u] - mean) * inv * g[c] + bta[c]);
  }
}

// ---------------------------------------------------------------------------
// Per-head projection (plain): Y[r, n*64+e] = sum_d X[r, n*64+d] * W[e,d]
// ---------------------------------------------------------------------------
__global__ void head_proj_kv(const __bf16* __restrict__ X, const __bf16* __restrict__ W,
                             __bf16* __restrict__ Y) {
  int wave = threadIdx.x >> 5, lane = threadIdx.x & 31;
  long r0 = (long)blockIdx.x * 64 + wave * 16;
  int n = blockIdx.y;
  const __bf16* a0 = X + r0 * kD + (long)n * kDH;
  v8f acc[4] = {};
  for (int kk = 0; kk < kDH; kk += 32) {
    v16bf a = load_frag(a0 + kk, kD, lane);
#pragma unroll
    for (int tI = 0; tI < 4; ++tI) {
      v16bf b = load_frag(W + (long)(16 * tI) * kDH + kk, kDH, lane);
      acc[tI] = wmma_bf16(a, b, acc[tI]);
    }
  }
  int half = lane >> 4, col = lane & 15;
#pragma unroll
  for (int tI = 0; tI < 4; ++tI)
#pragma unroll
    for (int v = 0; v < 8; ++v)
      Y[(r0 + v + 8 * half) * kD + (long)n * kDH + 16 * tI + col] = (__bf16)acc[tI][v];
}

// Query projection with dual relative-position biases
__global__ void head_proj_q(const __bf16* __restrict__ X, const __bf16* __restrict__ W,
                            const float* __restrict__ rwb, const float* __restrict__ rrb,
                            __bf16* __restrict__ Yrw, __bf16* __restrict__ Yrr) {
  int wave = threadIdx.x >> 5, lane = threadIdx.x & 31;
  long t0 = (long)blockIdx.x * 64 + wave * 16;   // flat (B*Q) output row
  int n = blockIdx.y;
  long bq = t0 / kQ, iq = t0 % kQ;
  long inr = bq * kK + kM + iq;                  // query rows inside vcat
  const __bf16* a0 = X + inr * kD + (long)n * kDH;
  v8f acc[4] = {};
  for (int kk = 0; kk < kDH; kk += 32) {
    v16bf a = load_frag(a0 + kk, kD, lane);
#pragma unroll
    for (int tI = 0; tI < 4; ++tI) {
      v16bf b = load_frag(W + (long)(16 * tI) * kDH + kk, kDH, lane);
      acc[tI] = wmma_bf16(a, b, acc[tI]);
    }
  }
  int half = lane >> 4, col = lane & 15;
#pragma unroll
  for (int tI = 0; tI < 4; ++tI) {
    int cb = n * kDH + 16 * tI + col;
    float bw = rwb[cb], br = rrb[cb];
#pragma unroll
    for (int v = 0; v < 8; ++v) {
      long oi = (t0 + v + 8 * half) * kD + (long)n * kDH + 16 * tI + col;
      Yrw[oi] = (__bf16)(acc[tI][v] + bw);
      Yrr[oi] = (__bf16)(acc[tI][v] + br);
    }
  }
}

// ---------------------------------------------------------------------------
// Generic flat GEMM: C(MxN) = A(MxKd) @ W(NxKd)^T ; flags: 1=accumulate, 2=relu
// Block tile 128(M)x64(N), 4 waves, wave tile 32x64.
// W tile (64x32 bf16) double-buffered through LDS via async global->LDS copies.
// Requires M % 128 == 0, N % 64 == 0, Kd % 32 == 0 (true for all uses here).
// ---------------------------------------------------------------------------
__global__ void gemm_xwt(const __bf16* __restrict__ A, int lda,
                         const __bf16* __restrict__ W, int ldw,
                         float* __restrict__ C, int ldc,
                         int M, int N, int Kd,
                         const float* __restrict__ bias, int flags) {
  __shared__ __align__(16) __bf16 Wt[2][64][40];   // padded rows: 80B stride
  int wave = threadIdx.x >> 5, lane = threadIdx.x & 31;
  int tid = threadIdx.x;
  int m0 = blockIdx.x * 128 + wave * 32;
  int n0 = blockIdx.y * 64;

  // stage a 64x32 W tile (k-offset kk) into LDS buffer `buf`
  auto stage = [&](int kk, int buf) {
#pragma unroll
    for (int it = 0; it < 2; ++it) {
      int c = tid + 128 * it;         // 256 chunks of 16B
      int row = c >> 2;
      int cc = (c & 3) * 8;
      async_copy16(W + (long)(n0 + row) * ldw + kk + cc, &Wt[buf][row][cc]);
    }
  };

  stage(0, 0);
  v8f acc[2][4] = {};
  int nk = Kd / 32;
  for (int ks = 0; ks < nk; ++ks) {
    int buf = ks & 1;
    wait_async_barrier();                    // stage ks landed; prev reads done
    if (ks + 1 < nk) stage((ks + 1) * 32, buf ^ 1);
    if (ks + 2 < nk) __builtin_prefetch(A + (long)m0 * lda + (ks + 2) * 32, 0, 3);
    v16bf a0 = load_frag(A + (long)m0 * lda + ks * 32, lda, lane);
    v16bf a1 = load_frag(A + (long)(m0 + 16) * lda + ks * 32, lda, lane);
#pragma unroll
    for (int tI = 0; tI < 4; ++tI) {
      v16bf b = load_frag_lds(&Wt[buf][16 * tI][0], lane);
      acc[0][tI] = wmma_bf16(a0, b, acc[0][tI]);
      acc[1][tI] = wmma_bf16(a1, b, acc[1][tI]);
    }
  }

  int half = lane >> 4, col = lane & 15;
#pragma unroll
  for (int mb = 0; mb < 2; ++mb)
#pragma unroll
    for (int tI = 0; tI < 4; ++tI) {
      int n = n0 + 16 * tI + col;
      float bv = bias ? bias[n] : 0.f;
#pragma unroll
      for (int v = 0; v < 8; ++v) {
        long ci = (long)(m0 + 16 * mb + v + 8 * half) * ldc + n;
        float x = acc[mb][tI][v] + bv;
        if (flags & 1) x += C[ci];
        if (flags & 2) x = fmaxf(x, 0.f);
        C[ci] = x;
      }
    }
}

// ---------------------------------------------------------------------------
// Per-(b,head) strided GEMM: z = blockIdx.z = b*H+n
//   C_z(MxN) = A_z(MxKd) @ B_z(NxKd)^T  with base = base0 + b*sb + n*sn
// ---------------------------------------------------------------------------
__global__ void gemm_bnh(const __bf16* __restrict__ A0, long Ab, long An, long Arow,
                         const __bf16* __restrict__ B0, long Bb, long Bn, long Brow,
                         float* __restrict__ C0, long Cb, long Cn, long Crow,
                         int M, int N, int Kd) {
  int z = blockIdx.z;
  int b = z / kH, n = z % kH;
  const __bf16* A = A0 + b * Ab + n * An;
  const __bf16* Bm = B0 + b * Bb + n * Bn;
  float* C = C0 + b * Cb + n * Cn;
  int wave = threadIdx.x >> 5, lane = threadIdx.x & 31;
  int m0 = blockIdx.x * 64 + wave * 16;
  int n0 = blockIdx.y * 64;
  if (m0 >= M) return;
  v8f acc[4] = {};
  for (int kk = 0; kk < Kd; kk += 32) {
    v16bf a = load_frag(A + (long)m0 * Arow + kk, Arow, lane);
#pragma unroll
    for (int tI = 0; tI < 4; ++tI) {
      int nn = n0 + 16 * tI;
      if (nn >= N) break;
      v16bf bf = load_frag(Bm + (long)nn * Brow + kk, Brow, lane);
      acc[tI] = wmma_bf16(a, bf, acc[tI]);
    }
  }
  int half = lane >> 4, col = lane & 15;
#pragma unroll
  for (int tI = 0; tI < 4; ++tI) {
    int nn = n0 + 16 * tI + col;
    if (nn >= N) continue;
#pragma unroll
    for (int v = 0; v < 8; ++v)
      C[(long)(m0 + v + 8 * half) * Crow + nn] = acc[tI][v];
  }
}

// ---------------------------------------------------------------------------
// Transformer-XL relative shift + scale + mask:
//   S[z,i,j] = (AC[z,i,j] + shift(BD)[z,i,j]) * SCALE ; masked -> -inf
// shift: xpf = i*K + j + Q ; qp = xpf/(K+1) ; kp = xpf%(K+1) ; 0 if kp==0
// ---------------------------------------------------------------------------
__global__ void shift_scale_mask(const float* __restrict__ AC, const float* __restrict__ BD,
                                 const unsigned char* __restrict__ mask,
                                 float* __restrict__ S) {
  int z = blockIdx.y;
  long e = (long)blockIdx.x * blockDim.x + threadIdx.x;
  if (e >= kQK) return;
  int i = (int)(e / kK), j = (int)(e % kK);
  int b = z / kH;
  long base = (long)z * kQK;
  long xpf = (long)i * kK + j + kQ;
  int qp = (int)(xpf / (kK + 1));
  int kp = (int)(xpf % (kK + 1));
  float bd = (kp == 0 || qp >= kQ) ? 0.f : BD[base + (long)qp * kK + (kp - 1)];
  float s = (AC[base + e] + bd) * kSCALE;
  if (mask[b * kK + j]) s = -__builtin_inff();
  S[base + e] = s;
}

// Row softmax over j; prob -> d_out in (i,j,b,n) order and bf16 copy for P@V
__global__ void softmax_row(const float* __restrict__ S, float* __restrict__ Pout,
                            __bf16* __restrict__ Pb) {
  int i = blockIdx.x, z = blockIdx.y, t = threadIdx.x;
  int b = z / kH, n = z % kH;
  const float* row = S + (long)z * kQK + (long)i * kK;
  __shared__ float red[256];
  float v[4];
  float mx = -__builtin_inff();
#pragma unroll
  for (int u = 0; u < 4; ++u) { v[u] = row[t + 256 * u]; mx = fmaxf(mx, v[u]); }
  red[t] = mx; __syncthreads();
  for (int o = 128; o > 0; o >>= 1) { if (t < o) red[t] = fmaxf(red[t], red[t + o]); __syncthreads(); }
  mx = red[0]; __syncthreads();
  float s = 0.f;
#pragma unroll
  for (int u = 0; u < 4; ++u) { v[u] = __expf(v[u] - mx); s += v[u]; }
  red[t] = s; __syncthreads();
  for (int o = 128; o > 0; o >>= 1) { if (t < o) red[t] += red[t + o]; __syncthreads(); }
  float inv = 1.f / red[0];
#pragma unroll
  for (int u = 0; u < 4; ++u) {
    int j = t + 256 * u;
    float pv = v[u] * inv;
    Pb[(long)z * kQK + (long)i * kK + j] = (__bf16)pv;
    Pout[(((long)i * kK + j) * kB + b) * kH + n] = pv;
  }
}

// ---------------------------------------------------------------------------
extern "C" void kernel_launch(void* const* d_in, const int* in_sizes, int n_in,
                              void* d_out, int out_size, void* d_ws, size_t ws_size,
                              hipStream_t stream) {
  (void)in_sizes; (void)n_in; (void)out_size; (void)ws_size;
  const float* value  = (const float*)d_in[0];
  const float* query  = (const float*)d_in[2];
  const unsigned char* mask = (const unsigned char*)d_in[3];
  const float* r      = (const float*)d_in[4];
  const float* rwb    = (const float*)d_in[5];
  const float* rrb    = (const float*)d_in[6];
  const float* Wv     = (const float*)d_in[7];
  const float* Wk     = (const float*)d_in[8];
  const float* Wq     = (const float*)d_in[9];
  const float* Wrnet  = (const float*)d_in[10];
  const float* ln1_g  = (const float*)d_in[11];
  const float* ln1_b  = (const float*)d_in[12];
  const float* ln2_g  = (const float*)d_in[13];
  const float* ln2_b  = (const float*)d_in[14];
  const float* lnkv_g = (const float*)d_in[15];
  const float* lnkv_b = (const float*)d_in[16];
  const float* fcW    = (const float*)d_in[17];
  const float* fcb    = (const float*)d_in[18];
  const float* g1[7]; for (int i = 0; i < 7; ++i) g1[i] = (const float*)d_in[19 + i];
  const float* g2[7]; for (int i = 0; i < 7; ++i) g2[i] = (const float*)d_in[26 + i];

  char* p = (char*)d_ws;
  auto alloc = [&](size_t bytes) { char* q = p; p += (bytes + 255) & ~(size_t)255; return q; };
  __bf16* vcat  = (__bf16*)alloc((size_t)kBK * kD * 2);
  __bf16* wv_s  = (__bf16*)alloc((size_t)kDH * kDH * 2);
  __bf16* wk_s  = (__bf16*)alloc((size_t)kDH * kDH * 2);
  __bf16* wq_s  = (__bf16*)alloc((size_t)kDH * kDH * 2);
  __bf16* wrn_b = (__bf16*)alloc((size_t)kD * kD * 2);
  __bf16* fcw_b = (__bf16*)alloc((size_t)kD * kD * 2);
  __bf16* gwb[12];
  for (int i = 0; i < 12; ++i) gwb[i] = (__bf16*)alloc((size_t)kD * kD * 2);
  __bf16* r_b   = (__bf16*)alloc((size_t)kK * kD * 2);
  float*  rk_f  = (float*)alloc((size_t)kK * kD * 4);
  __bf16* rk_b  = (__bf16*)alloc((size_t)kK * kD * 2);
  __bf16* wq_rw = (__bf16*)alloc((size_t)kBQ * kD * 2);
  __bf16* wq_rr = (__bf16*)alloc((size_t)kBQ * kD * 2);
  __bf16* wk_b  = (__bf16*)alloc((size_t)kBK * kD * 2);
  __bf16* wv_b  = (__bf16*)alloc((size_t)kBK * kD * 2);
  __bf16* wvT   = (__bf16*)alloc((size_t)kZH * kDH * kK * 2);
  float*  ACb   = (float*)alloc((size_t)kZH * kQK * 4);
  float*  BDb   = (float*)alloc((size_t)kZH * kQK * 4);
  __bf16* probb = (__bf16*)alloc((size_t)kZH * kQK * 2);
  float*  vec   = (float*)alloc((size_t)kBQ * kD * 4);
  __bf16* xb    = (__bf16*)alloc((size_t)kBQ * kD * 2);
  __bf16* yb    = (__bf16*)alloc((size_t)kBQ * kD * 2);
  float*  Tr    = (float*)alloc((size_t)kBQ * kD * 4);
  float*  Tz    = (float*)alloc((size_t)kBQ * kD * 4);
  float*  Th    = (float*)alloc((size_t)kBQ * kD * 4);
  __bf16* RX    = (__bf16*)alloc((size_t)kBQ * kD * 2);
  float*  hbuf  = (float*)alloc((size_t)kBQ * kD * 4);
  __bf16* hlnb  = (__bf16*)alloc((size_t)kBQ * kD * 2);
  float*  ff    = (float*)alloc((size_t)kBQ * kD * 4);

  float* out_f  = (float*)d_out;
  float* prob_f = out_f + kBQ * kD;

  auto cvt = [&](const float* s, __bf16* d, long n) {
    ew_cvt_bf16<<<dim3((unsigned)((n + 255) / 256)), dim3(256), 0, stream>>>(s, d, n);
  };

  // Weight + input conversions to bf16
  cvt(Wv, wv_s, kDH * kDH);
  cvt(Wk, wk_s, kDH * kDH);
  cvt(Wq, wq_s, kDH * kDH);
  cvt(Wrnet, wrn_b, (long)kD * kD);
  cvt(fcW, fcw_b, (long)kD * kD);
  for (int i = 0; i < 6; ++i) cvt(g1[i], gwb[i], (long)kD * kD);
  for (int i = 0; i < 6; ++i) cvt(g2[i], gwb[6 + i], (long)kD * kD);
  cvt(r, r_b, (long)kK * kD);
  cvt(query, xb, kBQ * kD);

  // LayerNorms -> v_cat (value rows, then query rows)
  ln_rows_bf16<<<dim3(kB * kM), dim3(256), 0, stream>>>(value, lnkv_g, lnkv_b, vcat, kM, kK, 0);
  ln_rows_bf16<<<dim3(kB * kQ), dim3(256), 0, stream>>>(query, ln1_g, ln1_b, vcat, kQ, kK, kM);

  // Per-head projections
  head_proj_kv<<<dim3((unsigned)(kBK / 64), kH), dim3(128), 0, stream>>>(vcat, wk_s, wk_b);
  head_proj_kv<<<dim3((unsigned)(kBK / 64), kH), dim3(128), 0, stream>>>(vcat, wv_s, wv_b);
  head_proj_q<<<dim3((unsigned)(kBQ / 64), kH), dim3(128), 0, stream>>>(vcat, wq_s, rwb, rrb, wq_rw, wq_rr);

  // r_k = r @ Wrnet^T
  gemm_xwt<<<dim3(kK / 128, kD / 64), dim3(128), 0, stream>>>(r_b, kD, wrn_b, kD, rk_f, kD,
                                                              kK, kD, kD, nullptr, 0);
  cvt(rk_f, rk_b, (long)kK * kD);

  // AC[z,i,j] = (wq+rw)[i*B+b, n] . wk[j*B+b, n]
  gemm_bnh<<<dim3(kQ / 64, kK / 64, kZH), dim3(128), 0, stream>>>(
      wq_rw, (long)kD, (long)kDH, (long)kB * kD,
      wk_b,  (long)kD, (long)kDH, (long)kB * kD,
      ACb,   (long)kH * kQK, (long)kQK, (long)kK,
      kQ, kK, kDH);
  // BDraw[z,i,j] = (wq+rr)[i*B+b, n] . r_k[j, n]
  gemm_bnh<<<dim3(kQ / 64, kK / 64, kZH), dim3(128), 0, stream>>>(
      wq_rr, (long)kD, (long)kDH, (long)kB * kD,
      rk_b,  0L, (long)kDH, (long)kD,
      BDb,   (long)kH * kQK, (long)kQK, (long)kK,
      kQ, kK, kDH);

  // rel-shift + scale + mask (in place into ACb), then softmax
  shift_scale_mask<<<dim3((unsigned)(kQK / 256), kZH), dim3(256), 0, stream>>>(ACb, BDb, mask, ACb);
  softmax_row<<<dim3(kQ, kZH), dim3(256), 0, stream>>>(ACb, prob_f, probb);

  // vec = prob @ w_v  (transpose w_v so its K dim is contiguous)
  transpose_wv<<<dim3((unsigned)(((long)kZH * kDH * kK + 255) / 256)), dim3(256), 0, stream>>>(wv_b, wvT);
  gemm_bnh<<<dim3(kQ / 64, 1, kZH), dim3(128), 0, stream>>>(
      probb, (long)kH * kQK, (long)kQK, (long)kK,
      wvT,   (long)kH * kDH * kK, (long)kDH * kK, (long)kK,
      vec,   (long)kD, (long)kDH, (long)kB * kD,
      kQ, kDH, kK);

  auto G = [&](const __bf16* A, const __bf16* W, float* C, const float* bias, int flags) {
    gemm_xwt<<<dim3((unsigned)(kBQ / 128), kD / 64), dim3(128), 0, stream>>>(
        A, kD, W, kD, C, kD, (int)kBQ, kD, kD, bias, flags);
  };
  unsigned ewg = (unsigned)((kBQ * kD + 255) / 256);

  // GRU gate 1: x = query, y = vec
  cvt(vec, yb, kBQ * kD);
  G(yb, gwb[0], Tr, nullptr, 0); G(xb, gwb[1], Tr, nullptr, 1);
  G(yb, gwb[2], Tz, nullptr, 0); G(xb, gwb[3], Tz, nullptr, 1);
  ew_sigmul_bf16<<<ewg, 256, 0, stream>>>(Tr, query, RX, kBQ * kD);
  G(yb, gwb[4], Th, nullptr, 0); G(RX, gwb[5], Th, nullptr, 1);
  ew_gru_out<<<ewg, 256, 0, stream>>>(query, Tz, Th, g1[6], hbuf, kBQ * kD);

  // ln2 + fc(relu)
  ln_rows_bf16<<<dim3((unsigned)kBQ), dim3(256), 0, stream>>>(hbuf, ln2_g, ln2_b, hlnb,
                                                              (int)kBQ, (int)kBQ, 0);
  G(hlnb, fcw_b, ff, fcb, 2);

  // GRU gate 2: x = h, y = ff ; output straight to d_out
  cvt(hbuf, xb, kBQ * kD);
  cvt(ff, yb, kBQ * kD);
  G(yb, gwb[6], Tr, nullptr, 0); G(xb, gwb[7], Tr, nullptr, 1);
  G(yb, gwb[8], Tz, nullptr, 0); G(xb, gwb[9], Tz, nullptr, 1);
  ew_sigmul_bf16<<<ewg, 256, 0, stream>>>(Tr, hbuf, RX, kBQ * kD);
  G(yb, gwb[10], Th, nullptr, 0); G(RX, gwb[11], Th, nullptr, 1);
  ew_gru_out<<<ewg, 256, 0, stream>>>(hbuf, Tz, Th, g2[6], out_f, kBQ * kD);
}
